// Experts_7705171329167
// MI455X (gfx1250) — matrix operand order repaired and checked by reading, not compiled, not run on hardware
//
#include <hip/hip_runtime.h>
#include <math.h>

// Problem constants (match reference)
#define DMODEL  2048
#define DFF     8192
#define NEXPERT 8
#define CAP     1024
#define NBATCH  4

typedef __bf16 bf16_t;
typedef __attribute__((ext_vector_type(16))) __bf16 v16bf;
typedef __attribute__((ext_vector_type(8)))  __bf16 v8bf;
typedef __attribute__((ext_vector_type(4)))  __bf16 v4bf;
typedef __attribute__((ext_vector_type(8)))  float  v8f;
typedef __attribute__((ext_vector_type(4)))  float  v4f;
typedef __attribute__((ext_vector_type(4)))  int    v4i;

#define BM 256
#define BN 128
#define BK 32
#define LDK (BK + 8)      // padded LDS row stride: 80 B (16B-aligned, de-strides banks)
#define THREADS 256       // 8 waves (wave32): 4 along M x 2 along N, 64x64 each

#define AS1 __attribute__((address_space(1)))
#define AS3 __attribute__((address_space(3)))

#if __has_builtin(__builtin_amdgcn_global_load_async_to_lds_b128) && \
    __has_builtin(__builtin_amdgcn_s_wait_asynccnt)
#define HAVE_ASYNC 1
#else
#define HAVE_ASYNC 0
#endif

// 16-byte global -> LDS copy (async DMA when available)
__device__ __forceinline__ void cp16(void* lds, const void* gmem) {
#if HAVE_ASYNC
    __builtin_amdgcn_global_load_async_to_lds_b128(
        (AS1 v4i*)gmem, (AS3 v4i*)lds, 0, 0);
#else
    *(v8bf*)lds = *(const v8bf*)gmem;
#endif
}

__device__ __forceinline__ float gelu_exact(float v) {
    return 0.5f * v * (1.0f + erff(v * 0.70710678118654752f));
}

// A fragment (16x32 bf16): lanes 0-15: K = 0..15; lanes 16-31: K = 16..31
__device__ __forceinline__ v16bf fragA_ld(const bf16_t* rowbase, int lane) {
    const int kb = (lane >= 16) ? 16 : 0;
    return *(const v16bf*)(rowbase + kb);
}

// B fragment (32x16 bf16, column per lane): lanes 0-15: K=0..15; lanes 16-31: K=16..31
__device__ __forceinline__ v16bf fragB_ld(const bf16_t* colbase, int lane) {
    const int kb = (lane >= 16) ? 16 : 0;
    return *(const v16bf*)(colbase + kb);
}

struct Tiles {
    bf16_t Ah[BM][LDK];
    bf16_t Al[BM][LDK];
    bf16_t Bh[BN][LDK];
    bf16_t Bl[BN][LDK];
};   // 60 KB

// Stage one BK-slice: pure byte copies (12 x 16B per thread = 12 async instr / wave)
template<int K>
__device__ __forceinline__ void stage(Tiles* t,
                                      const bf16_t* Ah, const bf16_t* Al,
                                      const bf16_t* Bh, const bf16_t* Bl, int tid) {
    #pragma unroll
    for (int i = 0; i < 4; ++i) {               // A: 256 rows x 64 B
        const int u = i * THREADS + tid;        // 0..1023
        const int row = u >> 2;
        const int part = (u & 3) * 8;
        cp16(&t->Ah[row][part], Ah + (size_t)row * K + part);
        cp16(&t->Al[row][part], Al + (size_t)row * K + part);
    }
    #pragma unroll
    for (int i = 0; i < 2; ++i) {               // B: 128 rows x 64 B
        const int u = i * THREADS + tid;        // 0..511
        const int row = u >> 2;
        const int part = (u & 3) * 8;
        cp16(&t->Bh[row][part], Bh + (size_t)row * K + part);
        cp16(&t->Bl[row][part], Bl + (size_t)row * K + part);
    }
}

// C = [gelu](A * B^T + bias):  A planes [chunks*CAP, K], B planes [E][N][K] (pre-transposed)
template<int N, int K, bool DO_GELU>
__global__ __launch_bounds__(THREADS)
void moe_gemm(const bf16_t* __restrict__ Agh, const bf16_t* __restrict__ Agl,
              const bf16_t* __restrict__ Bgh, const bf16_t* __restrict__ Bgl,
              const float*  __restrict__ bias,
              float* __restrict__ Cf, bf16_t* __restrict__ Ch, bf16_t* __restrict__ Cl)
{
    __shared__ Tiles tiles[2];                  // 120 KB of the 320 KB WGP pool

    const int tid   = threadIdx.x;
    const int lane  = tid & 31;
    const int wave  = tid >> 5;
    const int wm    = wave & 3;                 // 4 waves along M, 64 rows each
    const int wn    = wave >> 2;                // 2 waves along N, 64 cols each
    const int chunk = blockIdx.z;               // b*NEXPERT + e
    const int e     = chunk % NEXPERT;
    const int m0    = blockIdx.y * BM;
    const int n0    = blockIdx.x * BN;

    const bf16_t* Abh = Agh + (size_t)chunk * CAP * K + (size_t)m0 * K;
    const bf16_t* Abl = Agl + (size_t)chunk * CAP * K + (size_t)m0 * K;
    const bf16_t* Bbh = Bgh + (size_t)e * N * K + (size_t)n0 * K;
    const bf16_t* Bbl = Bgl + (size_t)e * N * K + (size_t)n0 * K;
    const float*  bb  = bias + (size_t)e * N + n0;

    v8f acc[4][4] = {};

    constexpr int nIter = K / BK;
    stage<K>(&tiles[0], Abh, Abl, Bbh, Bbl, tid);

    for (int it = 0; it < nIter; ++it) {
        if (it + 1 < nIter) {
            const int kn = (it + 1) * BK;
            stage<K>(&tiles[(it + 1) & 1], Abh + kn, Abl + kn, Bbh + kn, Bbl + kn, tid);
#if HAVE_ASYNC
            __builtin_amdgcn_s_wait_asynccnt(12);   // leave only next stage in flight
#endif
        } else {
#if HAVE_ASYNC
            __builtin_amdgcn_s_wait_asynccnt(0);
#endif
        }
        __syncthreads();

        Tiles* t = &tiles[it & 1];
        v16bf bh[4], bl[4];
        #pragma unroll
        for (int tn = 0; tn < 4; ++tn) {
            const int r = wn * 64 + tn * 16 + (lane & 15);
            bh[tn] = fragB_ld(&t->Bh[r][0], lane);
            bl[tn] = fragB_ld(&t->Bl[r][0], lane);
        }
        #pragma unroll
        for (int tm = 0; tm < 4; ++tm) {
            const int r = wm * 64 + tm * 16 + (lane & 15);
            const v16bf ah = fragA_ld(&t->Ah[r][0], lane);
            const v16bf al = fragA_ld(&t->Al[r][0], lane);
            #pragma unroll
            for (int tn = 0; tn < 4; ++tn) {
                acc[tm][tn] = __builtin_amdgcn_wmma_f32_16x16x32_bf16(
                    false, ah, false, bh[tn], (short)0, acc[tm][tn], false, false);
                acc[tm][tn] = __builtin_amdgcn_wmma_f32_16x16x32_bf16(
                    false, al, false, bh[tn], (short)0, acc[tm][tn], false, false);
                acc[tm][tn] = __builtin_amdgcn_wmma_f32_16x16x32_bf16(
                    false, ah, false, bl[tn], (short)0, acc[tm][tn], false, false);
            }
        }
        __syncthreads();
    }

    // Epilogue: C/D layout (lanes 0-15: N=lane, M=r; lanes 16-31: N=lane-16, M=8+r)
    const int nloc = lane & 15;
    const int mhi  = (lane >> 4) * 8;
    if constexpr (DO_GELU) {
        bf16_t* Chb = Ch + (size_t)chunk * CAP * N + (size_t)m0 * N + n0;
        bf16_t* Clb = Cl + (size_t)chunk * CAP * N + (size_t)m0 * N + n0;
        #pragma unroll
        for (int tm = 0; tm < 4; ++tm) {
            #pragma unroll
            for (int tn = 0; tn < 4; ++tn) {
                const int gn = wn * 64 + tn * 16 + nloc;
                const float bv = bb[gn];
                #pragma unroll
                for (int r = 0; r < 8; ++r) {
                    const int gm = wm * 64 + tm * 16 + mhi + r;
                    const float v = gelu_exact(acc[tm][tn][r] + bv);
                    const bf16_t h = (bf16_t)v;
                    const bf16_t l = (bf16_t)(v - (float)h);
                    Chb[(size_t)gm * N + gn] = h;
                    Clb[(size_t)gm * N + gn] = l;
                }
            }
        }
    } else {
        float* Cb = Cf + (size_t)chunk * CAP * N + (size_t)m0 * N + n0;
        #pragma unroll
        for (int tm = 0; tm < 4; ++tm) {
            #pragma unroll
            for (int tn = 0; tn < 4; ++tn) {
                const int gn = wn * 64 + tn * 16 + nloc;
                const float bv = bb[gn];
                #pragma unroll
                for (int r = 0; r < 8; ++r) {
                    const int gm = wm * 64 + tm * 16 + mhi + r;
                    Cb[(size_t)gm * N + gn] = acc[tm][tn][r] + bv;
                }
            }
        }
    }
}

// fp32 -> (bf16_hi, bf16_lo) planes, elementwise
__global__ __launch_bounds__(256)
void split_f32(const float* __restrict__ in, bf16_t* __restrict__ hi,
               bf16_t* __restrict__ lo, size_t n4) {
    size_t i = (size_t)blockIdx.x * blockDim.x + threadIdx.x;
    const size_t stride = (size_t)gridDim.x * blockDim.x;
    for (; i < n4; i += stride) {
        const v4f v = *(const v4f*)(in + i * 4);
        v4bf h, l;
        #pragma unroll
        for (int j = 0; j < 4; ++j) {
            h[j] = (bf16_t)v[j];
            l[j] = (bf16_t)(v[j] - (float)h[j]);
        }
        *(v4bf*)(hi + i * 4) = h;
        *(v4bf*)(lo + i * 4) = l;
    }
}

// w [E][K][N] fp32 -> wT hi/lo planes [E][N][K] bf16 (32x32 LDS tile transpose)
template<int K, int N>
__global__ __launch_bounds__(256)
void transpose_split(const float* __restrict__ in, bf16_t* __restrict__ hi,
                     bf16_t* __restrict__ lo) {
    __shared__ float tile[32][33];
    const int e  = blockIdx.z;
    const int n0 = blockIdx.x * 32;
    const int k0 = blockIdx.y * 32;
    const int tx = threadIdx.x & 31;
    const int ty = threadIdx.x >> 5;            // 0..7
    const float* src = in + (size_t)e * K * N;
    #pragma unroll
    for (int j = 0; j < 4; ++j)
        tile[ty + j * 8][tx] = src[(size_t)(k0 + ty + j * 8) * N + n0 + tx];
    __syncthreads();
    bf16_t* dh = hi + (size_t)e * N * K;
    bf16_t* dl = lo + (size_t)e * N * K;
    #pragma unroll
    for (int j = 0; j < 4; ++j) {
        const float f = tile[tx][ty + j * 8];
        const bf16_t h = (bf16_t)f;
        const bf16_t l = (bf16_t)(f - (float)h);
        dh[(size_t)(n0 + ty + j * 8) * K + k0 + tx] = h;
        dl[(size_t)(n0 + ty + j * 8) * K + k0 + tx] = l;
    }
}

extern "C" void kernel_launch(void* const* d_in, const int* in_sizes, int n_in,
                              void* d_out, int out_size, void* d_ws, size_t ws_size,
                              hipStream_t stream) {
    const float* x  = (const float*)d_in[0];  // [B, E*CAP, DMODEL]
    const float* w1 = (const float*)d_in[1];  // [E, DMODEL, DFF]
    const float* b1 = (const float*)d_in[2];  // [E, DFF]
    const float* w2 = (const float*)d_in[3];  // [E, DFF, DMODEL]
    const float* b2 = (const float*)d_in[4];  // [E, DMODEL]
    float* out = (float*)d_out;               // [B, E*CAP, DMODEL]

    // Workspace layout (bf16 elements): ~2.42 GB total
    const size_t Sx = (size_t)NBATCH * NEXPERT * CAP * DMODEL;  // 67,108,864
    const size_t Sw = (size_t)NEXPERT * DMODEL * DFF;           // 134,217,728
    const size_t Sh = (size_t)NBATCH * NEXPERT * CAP * DFF;     // 268,435,456
    bf16_t* Xh  = (bf16_t*)d_ws;
    bf16_t* Xl  = Xh  + Sx;
    bf16_t* W1h = Xl  + Sx;
    bf16_t* W1l = W1h + Sw;
    bf16_t* W2h = W1l + Sw;
    bf16_t* W2l = W2h + Sw;
    bf16_t* Hh  = W2l + Sw;
    bf16_t* Hl  = Hh  + Sh;

    // Preprocess: split x; transpose+split weights to [E][N][K]
    split_f32<<<4096, 256, 0, stream>>>(x, Xh, Xl, Sx / 4);
    transpose_split<DMODEL, DFF><<<dim3(DFF / 32, DMODEL / 32, NEXPERT), 256, 0, stream>>>(w1, W1h, W1l);
    transpose_split<DFF, DMODEL><<<dim3(DMODEL / 32, DFF / 32, NEXPERT), 256, 0, stream>>>(w2, W2h, W2l);

    // GEMM1 + GELU -> H (bf16 hi/lo planes)
    dim3 blk(THREADS);
    dim3 g1(DFF / BN, CAP / BM, NBATCH * NEXPERT);
    moe_gemm<DFF, DMODEL, true><<<g1, blk, 0, stream>>>(
        Xh, Xl, W1h, W1l, b1, nullptr, Hh, Hl);

    // GEMM2 -> out (fp32)
    dim3 g2(DMODEL / BN, CAP / BM, NBATCH * NEXPERT);
    moe_gemm<DMODEL, DFF, false><<<g2, blk, 0, stream>>>(
        Hh, Hl, W2h, W2l, b2, out, nullptr, nullptr);
}
